// FrequencyAwareSpatialAttention_21526376087981
// MI455X (gfx1250) — compile-verified, matrix operand order checked by reading, not run on hardware
//
#include <hip/hip_runtime.h>
#include <hip/hip_fp16.h>
#include <math.h>

typedef __attribute__((ext_vector_type(16))) _Float16 v16h;
typedef __attribute__((ext_vector_type(8)))  float    v8f;

#define WS     17      // window size
#define PW     19      // padded window (zero border for 3x3 conv)
#define C_IN   64
#define C_MID  16
#define NH     30
#define NW     30
#define HH     510
#define BATCH  4
#define NPIX   (WS*WS)        // 289
#define NTILE  19             // ceil(289/16)
#define THREADS 256
#define NWAVE  (THREADS/32)

#if defined(__has_builtin)
#if __has_builtin(__builtin_amdgcn_global_load_async_to_lds_b32) && \
    __has_builtin(__builtin_amdgcn_s_wait_asynccnt)
#define USE_ASYNC_LDS 1
#endif
#endif

#ifdef USE_ASYNC_LDS
typedef __attribute__((address_space(1))) int gas_int;   // prints as "__device__ int *"
typedef __attribute__((address_space(3))) int las_int;   // prints as "__shared__ int *"
#endif

union AU { v16h v; unsigned int u[8]; };

__global__ __launch_bounds__(THREADS)
void fasa_kernel(const float* __restrict__ x,
                 const float* __restrict__ sobel_w,   // [64,1,3,3]
                 const float* __restrict__ w1,        // [16,64,3,3]
                 const float* __restrict__ b1,        // [16]
                 const float* __restrict__ w2,        // [1,16,1,1]
                 const float* __restrict__ b2,        // [1]
                 float* __restrict__ out)             // [4,1,510,510]
{
    // LDS. xs is dead after phase 2 and is reused as per-wave reduction scratch.
    __shared__ float    xs[C_IN * PW * PW];            // 92416 B
    __shared__ _Float16 fq[PW * PW * C_IN];            // 46208 B
    __shared__ _Float16 bm[9 * C_MID * C_IN];          // 18432 B
    __shared__ float    sw[C_IN * 9];                  //  2304 B

    const int tid = threadIdx.x;
    const int blk = blockIdx.x;           // b*900 + wy*30 + wx
    const int b   = blk / (NH * NW);
    const int rem = blk % (NH * NW);
    const int wy  = rem / NW;
    const int wx  = rem % NW;

    // ---------------- Phase 1: stage inputs into LDS ----------------
    for (int j = tid; j < C_IN * 9; j += THREADS)
        sw[j] = sobel_w[j];

    // bm[(t*16 + n)*64 + c] = (f16) w1[n][c][t/3][t%3]
    for (int j = tid; j < 9 * C_MID * C_IN; j += THREADS) {
        int c = j & 63;
        int n = (j >> 6) & 15;
        int t = j >> 10;
        bm[j] = (_Float16)w1[(n * C_IN + c) * 9 + t];
    }

    // xs[c][yy][xx], zero border (yy,xx in [0,18], interior at +1)
    for (int j = tid; j < C_IN * PW * PW; j += THREADS) {
        int c  = j / (PW * PW);
        int r  = j % (PW * PW);
        int yy = r / PW, xx = r % PW;
        bool inb = (yy >= 1 && yy <= WS && xx >= 1 && xx <= WS);
#ifdef USE_ASYNC_LDS
        if (inb) {
            const float* gp = x + ((long)(b * C_IN + c) * HH + wy * WS + yy - 1) * HH
                                + wx * WS + xx - 1;
            __builtin_amdgcn_global_load_async_to_lds_b32(
                (gas_int*)gp, (las_int*)&xs[j], 0, 0);
        } else {
            xs[j] = 0.0f;
        }
#else
        float v = 0.0f;
        if (inb) {
            v = x[((long)(b * C_IN + c) * HH + wy * WS + yy - 1) * HH + wx * WS + xx - 1];
        }
        xs[j] = v;
#endif
    }
#ifdef USE_ASYNC_LDS
    __builtin_amdgcn_s_wait_asynccnt(0);
#endif
    __syncthreads();

    // ---------------- Phase 2: depthwise 3x3 sobel -> f16 freq ----------------
    // fq[(yy*19 + xx)*64 + c]; border cells written as 0 (conv zero-pad)
    for (int j = tid; j < PW * PW * C_IN; j += THREADS) {
        int c   = j & 63;
        int pix = j >> 6;
        int yy  = pix / PW, xx = pix % PW;
        float acc = 0.0f;
        if (yy >= 1 && yy <= WS && xx >= 1 && xx <= WS) {
            const float* xc  = xs + c * PW * PW;
            const float* swc = sw + c * 9;
            #pragma unroll
            for (int ky = 0; ky < 3; ++ky)
                #pragma unroll
                for (int kx = 0; kx < 3; ++kx)
                    acc += swc[ky * 3 + kx] * xc[(yy - 1 + ky) * PW + (xx - 1 + kx)];
        }
        fq[j] = (_Float16)acc;
    }
    __syncthreads();

    // ---------------- Phase 3: implicit-GEMM conv 64->16 via WMMA ----------------
    // M = pixel tile (16), N = 16 mid channels, K = 9 taps * 64 ch = 576 (18 x K32)
    const int wave = tid >> 5;
    const int lane = tid & 31;
    const int g    = lane >> 4;      // K-group select per ISA 16-bit layout
    const int n    = lane & 15;      // A-row (M) / B-col (N) index

    const float b1n = b1[n];
    const float w2n = w2[n];
    const float b2v = b2[0];

    // K offsets (half units) of the 8 dword pairs held by one lane (ISA 7.12.2)
    const int koff[8] = {0, 2, 4, 6, 16, 18, 20, 22};

    float* red = xs + wave * 256;    // per-wave 16x16 transpose scratch (reuses xs)

    for (int tile = wave; tile < NTILE; tile += NWAVE) {
        int p  = tile * 16 + n;
        int pc = p < NPIX ? p : NPIX - 1;  // clamped: WMMA rows are independent
        int py = pc / WS, px = pc % WS;

        // per-tile base pointers; every fragment offset below is a compile-time imm
        const char* a0 = (const char*)(fq + (py * PW + px) * C_IN + g * 8);
        const char* b0 = (const char*)(bm + n * C_IN + g * 8);

        v8f acc = {};
        #pragma unroll
        for (int kit = 0; kit < 18; ++kit) {
            const int t     = kit >> 1;            // tap 0..8
            const int chalf = (kit & 1) << 5;      // channel half: 0 or 32
            const int aoff  = ((t / 3) * PW + (t % 3)) * C_IN + chalf;
            const int boff  = t * C_MID * C_IN + chalf;
            AU a, bb;
            #pragma unroll
            for (int v = 0; v < 8; ++v) {
                a.u[v]  = *(const unsigned int*)(a0 + (aoff + koff[v]) * 2);
                bb.u[v] = *(const unsigned int*)(b0 + (boff + koff[v]) * 2);
            }
            acc = __builtin_amdgcn_wmma_f32_16x16x32_f16(
                      false, a.v, false, bb.v, (short)0, acc, false, false);
        }

        // Epilogue: +b1, ReLU, scale by w2, transpose via per-wave LDS scratch,
        // then 16 lanes reduce + sigmoid + clip + fold-write in parallel.
        #pragma unroll
        for (int r = 0; r < 8; ++r) {
            float h = acc[r] + b1n;
            h = fmaxf(h, 0.0f);
            red[(r + 8 * g) * 16 + n] = h * w2n;   // red[row][n]
        }
        __builtin_amdgcn_wave_barrier();           // intra-wave LDS ordering fence

        if (lane < 16) {
            const float4* rowp = (const float4*)(red + lane * 16);
            float4 q0 = rowp[0], q1 = rowp[1], q2 = rowp[2], q3 = rowp[3];
            float s = ((q0.x + q0.y) + (q0.z + q0.w))
                    + ((q1.x + q1.y) + (q1.z + q1.w))
                    + ((q2.x + q2.y) + (q2.z + q2.w))
                    + ((q3.x + q3.y) + (q3.z + q3.w));
            int prow = tile * 16 + lane;
            if (prow < NPIX) {
                float z   = s + b2v;
                float e   = __expf(-z);
                float sig = __builtin_amdgcn_rcpf(1.0f + e);
                sig = fminf(fmaxf(sig, 0.05f), 0.95f);
                int oy = prow / WS, ox = prow % WS;
                out[((long)b * HH + wy * WS + oy) * HH + wx * WS + ox] = sig;
            }
        }
        __builtin_amdgcn_wave_barrier();           // scratch reused next tile
    }
}

extern "C" void kernel_launch(void* const* d_in, const int* in_sizes, int n_in,
                              void* d_out, int out_size, void* d_ws, size_t ws_size,
                              hipStream_t stream) {
    const float* x  = (const float*)d_in[0];
    const float* sb = (const float*)d_in[1];
    const float* w1 = (const float*)d_in[2];
    const float* b1 = (const float*)d_in[3];
    const float* w2 = (const float*)d_in[4];
    const float* b2 = (const float*)d_in[5];
    float* out = (float*)d_out;

    dim3 grid(BATCH * NH * NW);   // 3600 windows, one block each
    fasa_kernel<<<grid, THREADS, 0, stream>>>(x, sb, w1, b1, w2, b2, out);

    (void)in_sizes; (void)n_in; (void)out_size; (void)d_ws; (void)ws_size;
}
